// CrossFusionAttention_30331059044905
// MI455X (gfx1250) — compile-verified, matrix operand order checked by reading, not run on hardware
//
#include <hip/hip_runtime.h>
#include <hip/hip_bf16.h>
#include <stdint.h>

typedef __attribute__((ext_vector_type(16))) _Float16 v16h;
typedef __attribute__((ext_vector_type(8)))  _Float16 v8h;
typedef __attribute__((ext_vector_type(2)))  _Float16 v2h;
typedef __attribute__((ext_vector_type(8)))  float    v8f;
typedef __attribute__((ext_vector_type(4)))  float    v4f;

#define BATCH 8
#define CH    64
#define CR    8
#define NPIX  4096   /* 64*64 */

// ---------------------------------------------------------------------------
// Kernel 1: fused 1x1 projections. Writes fragment-friendly f16 copies:
//   qT,kT : [B][N][8]  f16 (one 16B chunk per column -> one b128 per fragment)
//   vh    : [B][C][N]  f16 (row-contiguous V for async LDS staging)
//   vo    : [B][C][N]  f32 (for SE pooling + cross fusion in full precision)
// ---------------------------------------------------------------------------
__global__ __launch_bounds__(256) void cfa_qkv_proj(
    const float* __restrict__ x,
    const float* __restrict__ Wq, const float* __restrict__ bq,
    const float* __restrict__ Wk, const float* __restrict__ bk,
    const float* __restrict__ Wv, const float* __restrict__ bv,
    _Float16* __restrict__ qT, _Float16* __restrict__ kT,
    float* __restrict__ vo, _Float16* __restrict__ vh)
{
    const int n = blockIdx.x * 256 + threadIdx.x;   // 0..4095
    const int b = blockIdx.y;
    const float* xb = x + ((size_t)b * CH) * NPIX + n;

    float qa[CR], ka[CR], va[CH];
#pragma unroll
    for (int o = 0; o < CR; ++o) { qa[o] = bq[o]; ka[o] = bk[o]; }
#pragma unroll
    for (int o = 0; o < CH; ++o) { va[o] = bv[o]; }

    for (int c = 0; c < CH; ++c) {
        const float xc = xb[(size_t)c * NPIX];
#pragma unroll
        for (int o = 0; o < CR; ++o) {
            qa[o] += Wq[o * CH + c] * xc;
            ka[o] += Wk[o * CH + c] * xc;
        }
#pragma unroll
        for (int o = 0; o < CH; ++o) va[o] += Wv[o * CH + c] * xc;
    }

    v8h qh, kh;
#pragma unroll
    for (int o = 0; o < CR; ++o) {
        qh[o] = (_Float16)qa[o];
        kh[o] = (_Float16)ka[o];
    }
    *(v8h*)(qT + ((size_t)b * NPIX + n) * CR) = qh;
    *(v8h*)(kT + ((size_t)b * NPIX + n) * CR) = kh;

    float*    vb  = vo + ((size_t)b * CH) * NPIX + n;
    _Float16* vhb = vh + ((size_t)b * CH) * NPIX + n;
#pragma unroll
    for (int o = 0; o < CH; ++o) {
        vb[(size_t)o * NPIX]  = va[o];
        vhb[(size_t)o * NPIX] = (_Float16)va[o];
    }
}

// ---------------------------------------------------------------------------
// Kernel 2: flash-attention with WMMA (wave32).
// Block = 8 waves; wave owns 16 query rows (block: 128). Per 32-wide j tile:
//   - V tile staged to LDS with global_load_async_to_lds_b128 (ASYNCcnt)
//   - S = QK^T via 2x v_wmma_f32_16x16x32_f16 (K padded 8->32)
//   - online softmax in LDS (b128 reads), P stored f16
//   - O += P@V via 4x v_wmma, fragments gathered with ds b128 loads
// ---------------------------------------------------------------------------
__global__ __launch_bounds__(256) void cfa_attention(
    const _Float16* __restrict__ qT, const _Float16* __restrict__ kT,
    const _Float16* __restrict__ vh, float* __restrict__ self_out)
{
    __shared__ __align__(16) _Float16 sV[64][72];   // V tile [c][j], pitch 144B
    __shared__ __align__(16) float    sS[8][16][36];// per-wave scores, pitch 144B
    __shared__ __align__(16) _Float16 sP[8][16][40];// per-wave P (f16), pitch 80B
    __shared__ float sCorr[8][16];
    __shared__ float sL[8][16];

    const int tid    = threadIdx.x;
    const int wave   = tid >> 5;
    const int lane   = tid & 31;
    const int lane16 = lane & 15;
    const int half   = lane >> 4;              // 0: lanes 0-15, 1: lanes 16-31
    const int kbase  = half * 8;
    const int b      = blockIdx.y;
    const int iBase  = blockIdx.x * 128 + wave * 16;

    const _Float16* qTb = qT + (size_t)b * NPIX * CR;
    const _Float16* kTb = kT + (size_t)b * NPIX * CR;
    const _Float16* vhb = vh + (size_t)b * CH * NPIX;

    const v8h zero8 = (v8h)(_Float16)0.f;

    // Q A-fragment (f16 16x32 layout; only K=0..7 real, rest zero padding)
    v8h qlo = zero8;
    if (half == 0)
        qlo = *(const v8h*)(qTb + (size_t)(iBase + lane16) * CR);
    const v16h aq = __builtin_shufflevector(qlo, zero8,
        0, 1, 2, 3, 4, 5, 6, 7, 8, 9, 10, 11, 12, 13, 14, 15);

    // staging assignment: thread copies 16B of the [c][j] tile
    const int stC   = tid >> 2;               // 0..63
    const int stJ   = (tid & 3) * 8;          // 0,8,16,24
    const uint32_t ldsDst =
        (uint32_t)(uintptr_t)(&sV[stC][stJ]); // LDS byte offset (low 32 bits)

    v8f acc[4] = {v8f{}, v8f{}, v8f{}, v8f{}};
    float m_r = -1e30f, l_r = 0.f;

    for (int j0 = 0; j0 < NPIX; j0 += 32) {
        // ---- async stage of 32x64 V tile (f16) into LDS ----
        {
            const uint64_t gsrc = (uint64_t)(uintptr_t)
                (vhb + (size_t)stC * NPIX + j0 + stJ);
            asm volatile("global_load_async_to_lds_b128 %0, %1, off"
                         :: "v"(ldsDst), "v"(gsrc) : "memory");
        }

        // ---- S = Q K^T for two 16-col tiles (K fragment: one b128 each) ----
        v8h klo0 = zero8, klo1 = zero8;
        if (half == 0) {
            klo0 = *(const v8h*)(kTb + (size_t)(j0 + lane16) * CR);
            klo1 = *(const v8h*)(kTb + (size_t)(j0 + 16 + lane16) * CR);
        }
        const v16h bk0 = __builtin_shufflevector(klo0, zero8,
            0, 1, 2, 3, 4, 5, 6, 7, 8, 9, 10, 11, 12, 13, 14, 15);
        const v16h bk1 = __builtin_shufflevector(klo1, zero8,
            0, 1, 2, 3, 4, 5, 6, 7, 8, 9, 10, 11, 12, 13, 14, 15);

        v8f s0 = {}, s1 = {};
        s0 = __builtin_amdgcn_wmma_f32_16x16x32_f16(false, aq, false, bk0,
                                                    (short)0, s0, false, false);
        s1 = __builtin_amdgcn_wmma_f32_16x16x32_f16(false, aq, false, bk1,
                                                    (short)0, s1, false, false);
        // spill C-layout: VGPR r -> row r (lanes 0-15) / row r+8 (lanes 16-31)
#pragma unroll
        for (int r = 0; r < 8; ++r) {
            const int row = r + kbase;
            sS[wave][row][lane16]      = s0[r];
            sS[wave][row][16 + lane16] = s1[r];
        }
        asm volatile("s_wait_asynccnt 0x0" ::: "memory");
        __syncthreads();

        // ---- online softmax: lanes 0..15 own one row each ----
        if (lane < 16) {
            const int row = lane;
            float sv[32];
            float mx = -1e30f;
#pragma unroll
            for (int t4 = 0; t4 < 8; ++t4) {
                const v4f blk = *(const v4f*)&sS[wave][row][t4 * 4];
#pragma unroll
                for (int e = 0; e < 4; ++e) {
                    sv[t4 * 4 + e] = blk[e];
                    mx = fmaxf(mx, blk[e]);
                }
            }
            const float mnew = fmaxf(m_r, mx);
            const float corr = __expf(m_r - mnew);  // first iter: exp(-inf)=0
            float ps = 0.f;
#pragma unroll
            for (int c2 = 0; c2 < 32; c2 += 2) {
                const float p0 = __expf(sv[c2]     - mnew);
                const float p1 = __expf(sv[c2 + 1] - mnew);
                ps += p0 + p1;
                v2h pp;
                pp[0] = (_Float16)p0;
                pp[1] = (_Float16)p1;
                *(v2h*)&sP[wave][row][c2] = pp;
            }
            l_r = l_r * corr + ps;
            m_r = mnew;
            sCorr[wave][row] = corr;
        }
        __syncthreads();

        // ---- rescale accumulators, then O += P @ V ----
        float corrv[8];
#pragma unroll
        for (int r = 0; r < 8; ++r) corrv[r] = sCorr[wave][r + kbase];
#pragma unroll
        for (int ct = 0; ct < 4; ++ct)
#pragma unroll
            for (int r = 0; r < 8; ++r) acc[ct][r] *= corrv[r];

        // A fragment of P: two contiguous b128 LDS loads
        const v8h plo = *(const v8h*)&sP[wave][lane16][kbase];
        const v8h phi = *(const v8h*)&sP[wave][lane16][16 + kbase];
        const v16h ap = __builtin_shufflevector(plo, phi,
            0, 1, 2, 3, 4, 5, 6, 7, 8, 9, 10, 11, 12, 13, 14, 15);

        // B fragments of V^T: two contiguous b128 LDS loads per channel tile
#pragma unroll
        for (int ct = 0; ct < 4; ++ct) {
            const int c = ct * 16 + lane16;
            const v8h vlo = *(const v8h*)&sV[c][kbase];
            const v8h vhi = *(const v8h*)&sV[c][16 + kbase];
            const v16h bfrag = __builtin_shufflevector(vlo, vhi,
                0, 1, 2, 3, 4, 5, 6, 7, 8, 9, 10, 11, 12, 13, 14, 15);
            acc[ct] = __builtin_amdgcn_wmma_f32_16x16x32_f16(
                false, ap, false, bfrag, (short)0, acc[ct], false, false);
        }
        __syncthreads();   // protect sV/sS/sP before next tile overwrites
    }

    // ---- finalize: divide by softmax denominator, store [c][i] ----
    if (lane < 16) sL[wave][lane] = l_r;
    __syncthreads();
    float inv[8];
#pragma unroll
    for (int r = 0; r < 8; ++r) inv[r] = 1.0f / sL[wave][r + kbase];

    float* so = self_out + (size_t)b * CH * NPIX;
#pragma unroll
    for (int ct = 0; ct < 4; ++ct) {
#pragma unroll
        for (int r = 0; r < 8; ++r) {
            const int i = iBase + r + kbase;
            const int c = ct * 16 + lane16;
            so[(size_t)c * NPIX + i] = acc[ct][r] * inv[r];
        }
    }
}

// ---------------------------------------------------------------------------
// Kernel 3: SE branch. One block per batch.
// ---------------------------------------------------------------------------
__global__ __launch_bounds__(256) void cfa_se(
    const float* __restrict__ v, const float* __restrict__ Wfc1,
    const float* __restrict__ Wfc2, float* __restrict__ se_w)
{
    __shared__ float part[CH][5];
    __shared__ float pooledS[CH];
    __shared__ float hS[CR];

    const int b = blockIdx.x;
    const int t = threadIdx.x;
    const int c = t & 63;
    const int chunk = t >> 6;                 // 0..3
    const float* vb = v + ((size_t)b * CH + c) * NPIX;

    float s = 0.f;
    for (int n = chunk; n < NPIX; n += 4) s += vb[n];
    part[c][chunk] = s;
    __syncthreads();

    if (t < CH) {
        pooledS[t] = (part[t][0] + part[t][1] + part[t][2] + part[t][3]) *
                     (1.0f / (float)NPIX);
    }
    __syncthreads();
    if (t < CR) {
        float h = 0.f;
        for (int cc = 0; cc < CH; ++cc) h += pooledS[cc] * Wfc1[t * CH + cc];
        hS[t] = fmaxf(h, 0.f);
    }
    __syncthreads();
    if (t < CH) {
        float z = 0.f;
#pragma unroll
        for (int o = 0; o < CR; ++o) z += hS[o] * Wfc2[t * CR + o];
        se_w[b * CH + t] = 1.0f / (1.0f + __expf(-z));
    }
}

// ---------------------------------------------------------------------------
// Kernel 4: cross fusion 1x1 conv on concat([self_out, v*se_w]).
// ---------------------------------------------------------------------------
__global__ __launch_bounds__(256) void cfa_cross(
    const float* __restrict__ self_out, const float* __restrict__ v,
    const float* __restrict__ se_w, const float* __restrict__ Wc,
    const float* __restrict__ bc, float* __restrict__ cross_feat)
{
    const int n = blockIdx.x * 256 + threadIdx.x;
    const int b = blockIdx.y;

    float accv[CH];
#pragma unroll
    for (int o = 0; o < CH; ++o) accv[o] = bc[o];

    for (int c = 0; c < CH; ++c) {
        if (c + 1 < CH) {
            __builtin_prefetch(&self_out[((size_t)b * CH + c + 1) * NPIX + n], 0, 1);
            __builtin_prefetch(&v[((size_t)b * CH + c + 1) * NPIX + n], 0, 1);
        }
        const float so = self_out[((size_t)b * CH + c) * NPIX + n];
        const float vv = v[((size_t)b * CH + c) * NPIX + n] * se_w[b * CH + c];
#pragma unroll
        for (int o = 0; o < CH; ++o)
            accv[o] += Wc[o * (2 * CH) + c] * so + Wc[o * (2 * CH) + CH + c] * vv;
    }
#pragma unroll
    for (int o = 0; o < CH; ++o)
        cross_feat[((size_t)b * CH + o) * NPIX + n] = accv[o];
}

// ---------------------------------------------------------------------------
// Kernel 5: 3x3 SAME conv + residual combine.
// out = x + gamma*self_out + beta*(conv3x3(cross_feat) + bf)
// ---------------------------------------------------------------------------
__global__ __launch_bounds__(256) void cfa_fuse(
    const float* __restrict__ x, const float* __restrict__ cross_feat,
    const float* __restrict__ Wf, const float* __restrict__ bf,
    const float* __restrict__ self_out, const float* __restrict__ gamma,
    const float* __restrict__ beta, float* __restrict__ out)
{
    const int tid = threadIdx.x;
    const int w  = tid & 63;
    const int h  = blockIdx.x * 4 + (tid >> 6);
    const int co = blockIdx.y;
    const int b  = blockIdx.z;

    float accs = bf[co];
    for (int ci = 0; ci < CH; ++ci) {
        const float* cf = cross_feat + ((size_t)b * CH + ci) * NPIX;
        if (ci + 1 < CH) __builtin_prefetch(cf + NPIX + h * 64 + w, 0, 1);
        const float* wt = Wf + ((size_t)co * CH + ci) * 9;
#pragma unroll
        for (int kh = 0; kh < 3; ++kh) {
            const int hy = h + kh - 1;
            if (hy < 0 || hy >= 64) continue;
#pragma unroll
            for (int kw = 0; kw < 3; ++kw) {
                const int wx = w + kw - 1;
                if (wx < 0 || wx >= 64) continue;
                accs += wt[kh * 3 + kw] * cf[hy * 64 + wx];
            }
        }
    }
    const size_t idx = ((size_t)b * CH + co) * NPIX + h * 64 + w;
    out[idx] = x[idx] + gamma[0] * self_out[idx] + beta[0] * accs;
}

// ---------------------------------------------------------------------------
extern "C" void kernel_launch(void* const* d_in, const int* in_sizes, int n_in,
                              void* d_out, int out_size, void* d_ws, size_t ws_size,
                              hipStream_t stream)
{
    const float* x     = (const float*)d_in[0];
    const float* Wq    = (const float*)d_in[1];
    const float* bq    = (const float*)d_in[2];
    const float* Wk    = (const float*)d_in[3];
    const float* bk    = (const float*)d_in[4];
    const float* Wv    = (const float*)d_in[5];
    const float* bv    = (const float*)d_in[6];
    const float* Wfc1  = (const float*)d_in[7];
    const float* Wfc2  = (const float*)d_in[8];
    const float* Wc    = (const float*)d_in[9];
    const float* bc    = (const float*)d_in[10];
    const float* Wf    = (const float*)d_in[11];
    const float* bf    = (const float*)d_in[12];
    const float* gamma = (const float*)d_in[13];
    const float* beta  = (const float*)d_in[14];
    float* out = (float*)d_out;

    // workspace layout (bytes; every section 16B-aligned by construction)
    char* ws = (char*)d_ws;
    float*    v_ws  = (float*)ws;                                  // 8 MB
    float*    so_ws = (float*)(ws + (size_t)8 * 1024 * 1024);      // 8 MB
    float*    cf_ws = (float*)(ws + (size_t)16 * 1024 * 1024);     // 8 MB
    _Float16* qT_ws = (_Float16*)(ws + (size_t)24 * 1024 * 1024);  // 512 KB
    _Float16* kT_ws = (_Float16*)(ws + (size_t)24 * 1024 * 1024 + 512 * 1024);
    _Float16* vh_ws = (_Float16*)(ws + (size_t)25 * 1024 * 1024);  // 4 MB
    float*    sw_ws = (float*)(ws + (size_t)29 * 1024 * 1024);     // 2 KB

    cfa_qkv_proj<<<dim3(NPIX / 256, BATCH), 256, 0, stream>>>(
        x, Wq, bq, Wk, bk, Wv, bv, qT_ws, kT_ws, v_ws, vh_ws);

    cfa_attention<<<dim3(NPIX / 128, BATCH), 256, 0, stream>>>(
        qT_ws, kT_ws, vh_ws, so_ws);

    cfa_se<<<dim3(BATCH), 256, 0, stream>>>(v_ws, Wfc1, Wfc2, sw_ws);

    cfa_cross<<<dim3(NPIX / 256, BATCH), 256, 0, stream>>>(
        so_ws, v_ws, sw_ws, Wc, bc, cf_ws);

    cfa_fuse<<<dim3(16, CH, BATCH), 256, 0, stream>>>(
        x, cf_ws, Wf, bf, so_ws, gamma, beta, out);
}
